// Linear_86801289052264
// MI455X (gfx1250) — 8-bit matrix operand order repaired and checked by reading, compiled, not run on hardware
//
#include <hip/hip_runtime.h>
#include <stdint.h>

typedef __attribute__((ext_vector_type(16))) int   v16i;
typedef __attribute__((ext_vector_type(8)))  float v8f;

#define M_DIM 8192
#define K_DIM 4096
#define N_DIM 14336

#define BM 128
#define BN 128
#define BK 128
#define LDS_STRIDE 144
#define TILE_BYTES (BM * LDS_STRIDE)
#define NKB (K_DIM / BK)

typedef __attribute__((address_space(3))) uint8_t       lds_u8;
typedef __attribute__((address_space(1))) const uint8_t glb_u8;

__device__ __forceinline__ void async_copy16(const uint8_t* g, uint8_t* l) {
#if __has_builtin(__builtin_amdgcn_global_load_async_to_lds_b128)
    typedef int v4i_t __attribute__((vector_size(16)));
    typedef __attribute__((address_space(1))) v4i_t gv4;
    typedef __attribute__((address_space(3))) v4i_t lv4;
    __builtin_amdgcn_global_load_async_to_lds_b128((gv4*)g, (lv4*)l, 0, 0);
#else
    unsigned lofs = (unsigned)(size_t)(lds_u8*)l;
    asm volatile("global_load_async_to_lds_b128 %0, %1, off"
                 :: "v"(lofs), "v"(g) : "memory");
#endif
}

template <int N>
__device__ __forceinline__ void wait_asynccnt() {
#if __has_builtin(__builtin_amdgcn_s_wait_asynccnt)
    __builtin_amdgcn_s_wait_asynccnt(N);
#else
    asm volatile("s_wait_asynccnt %0" :: "i"(N) : "memory");
#endif
}

__device__ __forceinline__ unsigned int f32_to_e4m3(float x) {
    x = fminf(448.0f, fmaxf(-448.0f, x));
    unsigned int u    = __float_as_uint(x);
    unsigned int sign = (u >> 24) & 0x80u;
    unsigned int a    = u & 0x7fffffffu;
    unsigned int r;
    if (a >= 0x3c800000u) {
        unsigned int rnd = a + 0x7ffffu + ((a >> 20) & 1u);
        unsigned int e   = (rnd >> 23) - 127u + 7u;
        r = (e << 3) | ((rnd >> 20) & 7u);
    } else {
        r = (unsigned int)(int)rintf(__uint_as_float(a) * 512.0f);
    }
    return sign | r;
}

__device__ __forceinline__ unsigned int pack4_e4m3(float a, float b, float c, float d) {
    return f32_to_e4m3(a) | (f32_to_e4m3(b) << 8) |
           (f32_to_e4m3(c) << 16) | (f32_to_e4m3(d) << 24);
}

__global__ void quantize_e4m3_kernel(const float* __restrict__ src,
                                     uint8_t* __restrict__ dst,
                                     long n16,
                                     const float* __restrict__ scale,
                                     int apply_scale) {
    long i = (long)blockIdx.x * blockDim.x + threadIdx.x;
    if (i >= n16) return;
    float sc = apply_scale ? scale[0] : 1.0f;
    const float4* s4 = (const float4*)(src + i * 16);
    unsigned int o[4];
#pragma unroll
    for (int j = 0; j < 4; ++j) {
        float4 v = s4[j];
        if (apply_scale) { v.x /= sc; v.y /= sc; v.z /= sc; v.w /= sc; }
        o[j] = pack4_e4m3(v.x, v.y, v.z, v.w);
    }
    *(uint4*)(dst + i * 16) = make_uint4(o[0], o[1], o[2], o[3]);
}

__global__ __launch_bounds__(256)
void gemm_fp8_wmma_kernel(const uint8_t* __restrict__ Aq,
                          const uint8_t* __restrict__ Bq,
                          const float*   __restrict__ bias,
                          const float*   __restrict__ s_in,
                          const float*   __restrict__ s_w,
                          float*         __restrict__ out) {
    __shared__ __align__(16) uint8_t lA[2 * TILE_BYTES];
    __shared__ __align__(16) uint8_t lB[2 * TILE_BYTES];

    const int tid   = threadIdx.x;
    const int lane  = tid & 31;
    const int wave  = tid >> 5;
    const int wr    = wave >> 2;
    const int wc    = wave & 3;
    const int laneN = lane & 15;
    const int laneH = lane >> 4;

    const long blockM = (long)blockIdx.y * BM;
    const long blockN = (long)blockIdx.x * BN;

    const int gRow = tid >> 3;
    const int gCol = (tid & 7) * 16;

    const uint8_t* gA = Aq + (blockM + gRow) * (long)K_DIM + gCol;
    const uint8_t* gB = Bq + (blockN + gRow) * (long)K_DIM + gCol;
    uint8_t* sA = lA + gRow * LDS_STRIDE + gCol;
    uint8_t* sB = lB + gRow * LDS_STRIDE + gCol;

    v8f acc[4][2];
#pragma unroll
    for (int mt = 0; mt < 4; ++mt)
#pragma unroll
        for (int nt = 0; nt < 2; ++nt)
            acc[mt][nt] = (v8f){0.f, 0.f, 0.f, 0.f, 0.f, 0.f, 0.f, 0.f};

#pragma unroll
    for (int r = 0; r < BM; r += 32) {
        async_copy16(gA + r * (long)K_DIM, sA + r * LDS_STRIDE);
        async_copy16(gB + r * (long)K_DIM, sB + r * LDS_STRIDE);
    }

    int buf = 0;
    for (int kb = 0; kb < NKB; ++kb) {
        const bool has_next = (kb + 1) < NKB;
        if (has_next) {
            const long knext = (long)(kb + 1) * BK;
            const int  bo    = (buf ^ 1) * TILE_BYTES;
#pragma unroll
            for (int r = 0; r < BM; r += 32) {
                async_copy16(gA + r * (long)K_DIM + knext, sA + r * LDS_STRIDE + bo);
                async_copy16(gB + r * (long)K_DIM + knext, sB + r * LDS_STRIDE + bo);
            }
            wait_asynccnt<8>();
        } else {
            wait_asynccnt<0>();
        }
        __syncthreads();

        const uint8_t* cA = lA + buf * TILE_BYTES;
        const uint8_t* cB = lB + buf * TILE_BYTES;

        v16i afrag[4];
#pragma unroll
        for (int mt = 0; mt < 4; ++mt) {
            const uint8_t* base = cA + (wr * 64 + mt * 16 + laneN) * LDS_STRIDE + laneH * 8;
#pragma unroll
            for (int i = 0; i < 8; ++i) {
                uint2 d = *(const uint2*)(base + i * 16);
                afrag[mt][2 * i]     = (int)d.x;
                afrag[mt][2 * i + 1] = (int)d.y;
            }
        }
        v16i bfrag[2];
#pragma unroll
        for (int nt = 0; nt < 2; ++nt) {
            const uint8_t* base = cB + (wc * 32 + nt * 16 + laneN) * LDS_STRIDE + laneH * 8;
#pragma unroll
            for (int i = 0; i < 4; ++i) {
                uint2 d0 = *(const uint2*)(base + i * 32);
                uint2 d1 = *(const uint2*)(base + i * 32 + 16);
                bfrag[nt][4 * i + 0] = (int)d0.x;
                bfrag[nt][4 * i + 1] = (int)d0.y;
                bfrag[nt][4 * i + 2] = (int)d1.x;
                bfrag[nt][4 * i + 3] = (int)d1.y;
            }
        }

#pragma unroll
        for (int mt = 0; mt < 4; ++mt)
#pragma unroll
            for (int nt = 0; nt < 2; ++nt)
                acc[mt][nt] = __builtin_amdgcn_wmma_f32_16x16x128_fp8_fp8(
                    afrag[mt], bfrag[nt], (short)0, acc[mt][nt],
                     false,  false);

        __syncthreads();
        buf ^= 1;
    }

    const float s = s_in[0] * s_w[0];
#pragma unroll
    for (int mt = 0; mt < 4; ++mt) {
#pragma unroll
        for (int nt = 0; nt < 2; ++nt) {
            const long n = blockN + wc * 32 + nt * 16 + laneN;
            const float b = bias[n];
            const long mBase = blockM + wr * 64 + mt * 16 + laneH * 8;
#pragma unroll
            for (int v = 0; v < 8; ++v)
                out[(mBase + v) * (long)N_DIM + n] = acc[mt][nt][v] * s + b;
        }
    }
}

extern "C" void kernel_launch(void* const* d_in, const int* in_sizes, int n_in,
                              void* d_out, int out_size, void* d_ws, size_t ws_size,
                              hipStream_t stream) {
    const float* x      = (const float*)d_in[0];
    const float* weight = (const float*)d_in[1];
    const float* bias   = (const float*)d_in[2];
    const float* s_in   = (const float*)d_in[3];
    const float* s_w    = (const float*)d_in[4];
    float*       out    = (float*)d_out;

    uint8_t* xq = (uint8_t*)d_ws;
    uint8_t* wq = xq + (size_t)M_DIM * K_DIM;

    const long nx16 = (long)M_DIM * K_DIM / 16;
    const long nw16 = (long)N_DIM * K_DIM / 16;

    quantize_e4m3_kernel<<<dim3((unsigned)((nx16 + 255) / 256)), dim3(256), 0, stream>>>(
        x, xq, nx16, s_in, 1);
    quantize_e4m3_kernel<<<dim3((unsigned)((nw16 + 255) / 256)), dim3(256), 0, stream>>>(
        weight, wq, nw16, s_in, 0);

    dim3 grid(N_DIM / BN, M_DIM / BM);
    gemm_fp8_wmma_kernel<<<grid, dim3(256), 0, stream>>>(xq, wq, bias, s_in, s_w, out);
}
